// AutoregressiveSender_34333968564758
// MI455X (gfx1250) — matrix operand order repaired and checked by reading, not compiled, not run on hardware
//
#include <hip/hip_runtime.h>
#include <hip/hip_bf16.h>
#include <stdint.h>

// ---------------- problem constants ----------------
static constexpr int Bn  = 8192;   // batch
static constexpr int DIN = 2048;   // encoder input
static constexpr int Hd  = 1024;   // hidden
static constexpr int Vn  = 6;      // vocab
static constexpr int Tn  = 6;      // steps
static constexpr int Kd  = 2048;   // GEMM K (== 2H == DIN)

typedef __bf16 bf16;
typedef __attribute__((ext_vector_type(16))) __bf16 v16bf;
typedef __attribute__((ext_vector_type(8)))  __bf16 v8bf;
typedef __attribute__((ext_vector_type(8)))  float  v8f;
typedef __attribute__((ext_vector_type(4)))  int    v4i;

union FragU { v16bf v; v8bf h[2]; };

// CDNA5 async global->LDS copy path (probe via __has_builtin; safe fallback).
#if defined(__has_builtin)
#if __has_builtin(__builtin_amdgcn_global_load_async_to_lds_b128) && \
    __has_builtin(__builtin_amdgcn_s_wait_asynccnt)
#define USE_ASYNC_LDS 1
#endif
#endif
#ifndef USE_ASYNC_LDS
#define USE_ASYNC_LDS 0
#endif

static __device__ __forceinline__ bf16 f2bf(float f) {
    union { float f; uint32_t u; } a; a.f = f;
    uint32_t r = a.u + 0x7FFFu + ((a.u >> 16) & 1u);   // RNE
    union { uint16_t s; bf16 b; } o; o.s = (uint16_t)(r >> 16);
    return o.b;
}

#if USE_ASYNC_LDS
typedef __attribute__((address_space(1))) v4i glob_v4i;
typedef __attribute__((address_space(3))) v4i  lds_v4i;
static __device__ __forceinline__ void cp16_async(bf16* lds_dst, const bf16* gsrc) {
    // global generic pointers are value-identical to AS1; LDS needs addrspacecast
    __builtin_amdgcn_global_load_async_to_lds_b128(
        (glob_v4i*)(uintptr_t)gsrc,
        (lds_v4i*)lds_dst, 0, 0);
}
#endif

// ---------------- conversion kernels ----------------
__global__ __launch_bounds__(256) void k_cvt_bf16(const float* __restrict__ in,
                                                  bf16* __restrict__ out, int n) {
    int i = blockIdx.x * 256 + threadIdx.x;
    if (i < n) out[i] = f2bf(in[i]);
}

// in: (K,N) f32 row-major  ->  out: (N,K) bf16 row-major
__global__ __launch_bounds__(256) void k_transpose_bf16(const float* __restrict__ in,
                                                        bf16* __restrict__ out,
                                                        int K, int N) {
    __shared__ float tile[32][33];
    int k0 = blockIdx.x * 32, n0 = blockIdx.y * 32;
    for (int i = threadIdx.y; i < 32; i += 8)
        tile[i][threadIdx.x] = in[(size_t)(k0 + i) * N + n0 + threadIdx.x];
    __syncthreads();
    for (int i = threadIdx.y; i < 32; i += 8)
        out[(size_t)(n0 + i) * K + k0 + threadIdx.x] = f2bf(tile[threadIdx.x][i]);
}

// comb_bf[:,0:H] = bf16(start_embed), rc=1, nt=0
__global__ __launch_bounds__(256) void k_init(const float* __restrict__ start_embed,
                                              bf16* __restrict__ comb_bf,
                                              float* __restrict__ rc,
                                              float* __restrict__ nt) {
    int i = blockIdx.x * 256 + threadIdx.x;            // over B*H
    int b = i >> 10, c = i & (Hd - 1);
    comb_bf[(size_t)b * (2 * Hd) + c] = f2bf(start_embed[c]);
    if (c == 0) { rc[b] = 1.0f; nt[b] = 0.0f; }
}

// ---------------- WMMA GEMM: out = epilogue(A @ W^T + bias) ----------------
// A is split: k < ksplit -> A0 (lda0), else A1 (lda1).  WT is (N=1024, K) bf16.
// Workgroup tile 128x128, 8 waves, wave tile 64x32, K-chunk 32, double-buffered.
// MODE 0: enc  -> h = acc+bias;           comb_bf[:,H+n] = bf16(h)
// MODE 1: z    -> z = sigmoid(acc+bias)
// MODE 2: r    -> rh_bf = bf16(sigmoid(acc+bias) * h)
// MODE 3: h~   -> hn=(1-z)*h+z*tanh(acc+bias); h=hn; comb_bf[:,H+n]=bf16(hn)
template <int MODE>
__global__ __launch_bounds__(256)
void gemm_bf16(const bf16* __restrict__ A0, int lda0,
               const bf16* __restrict__ A1, int lda1, int ksplit,
               const bf16* __restrict__ WT, int K,
               const float* __restrict__ bias,
               float* __restrict__ hbuf, float* __restrict__ zbuf,
               bf16* __restrict__ comb_bf, bf16* __restrict__ rh_bf) {
    constexpr int LDT = 40;                               // padded row (bf16), 80B
    __shared__ __attribute__((aligned(16))) bf16 Alds[2][128 * LDT];
    __shared__ __attribute__((aligned(16))) bf16 Blds[2][128 * LDT];

    const int tid  = threadIdx.x;
    const int lane = tid & 31;
    const int wave = tid >> 5;
    const int wy   = wave >> 2;          // 0..1  -> 64-row slice
    const int wx   = wave & 3;           // 0..3  -> 32-col slice
    const int m0   = blockIdx.x * 128;
    const int n0   = blockIdx.y * 128;
    const int half = lane >> 4;
    const int l15  = lane & 15;

    // per-thread staging slots: q = 16B segment, rows r0 and r0+64
    const int q  = tid & 3;
    const int r0 = tid >> 2;             // 0..63
    const int r1 = r0 + 64;

    // global source row pointers for a given k-chunk
    auto Arow = [&](int r, int kk) -> const bf16* {
        return (kk < ksplit)
            ? A0 + (size_t)(m0 + r) * lda0 + kk + q * 8
            : A1 + (size_t)(m0 + r) * lda1 + (kk - ksplit) + q * 8;
    };
    auto Brow = [&](int r, int kk) -> const bf16* {
        return WT + (size_t)(n0 + r) * K + kk + q * 8;
    };

    v8f acc[4][2] = {};

#if USE_ASYNC_LDS
    // -------- async double-buffered pipeline --------
    cp16_async(&Alds[0][r0 * LDT + q * 8], Arow(r0, 0));
    cp16_async(&Alds[0][r1 * LDT + q * 8], Arow(r1, 0));
    cp16_async(&Blds[0][r0 * LDT + q * 8], Brow(r0, 0));
    cp16_async(&Blds[0][r1 * LDT + q * 8], Brow(r1, 0));
    __builtin_amdgcn_s_wait_asynccnt(0);
    __syncthreads();
#else
    // -------- register-staged double-buffered pipeline --------
    {
        v8bf ra0 = *(const v8bf*)Arow(r0, 0);
        v8bf ra1 = *(const v8bf*)Arow(r1, 0);
        v8bf rb0 = *(const v8bf*)Brow(r0, 0);
        v8bf rb1 = *(const v8bf*)Brow(r1, 0);
        *(v8bf*)&Alds[0][r0 * LDT + q * 8] = ra0;
        *(v8bf*)&Alds[0][r1 * LDT + q * 8] = ra1;
        *(v8bf*)&Blds[0][r0 * LDT + q * 8] = rb0;
        *(v8bf*)&Blds[0][r1 * LDT + q * 8] = rb1;
        __syncthreads();
    }
#endif

    for (int kk = 0; kk < K; kk += 32) {
        const int cur = (kk >> 5) & 1;
        const int nxt = cur ^ 1;
        const bool more = (kk + 32) < K;

#if USE_ASYNC_LDS
        if (more) {   // fetch next chunk while computing this one
            cp16_async(&Alds[nxt][r0 * LDT + q * 8], Arow(r0, kk + 32));
            cp16_async(&Alds[nxt][r1 * LDT + q * 8], Arow(r1, kk + 32));
            cp16_async(&Blds[nxt][r0 * LDT + q * 8], Brow(r0, kk + 32));
            cp16_async(&Blds[nxt][r1 * LDT + q * 8], Brow(r1, kk + 32));
        }
#else
        v8bf ra0, ra1, rb0, rb1;
        if (more) {   // issue loads early; consumed only after the WMMAs
            ra0 = *(const v8bf*)Arow(r0, kk + 32);
            ra1 = *(const v8bf*)Arow(r1, kk + 32);
            rb0 = *(const v8bf*)Brow(r0, kk + 32);
            rb1 = *(const v8bf*)Brow(r1, kk + 32);
        }
#endif

        // A fragments: lane<16 holds K[0..7],K[16..23]; lane>=16 K[8..15],K[24..31]
        FragU a[4];
        #pragma unroll
        for (int i = 0; i < 4; ++i) {
            const bf16* p = &Alds[cur][(wy * 64 + i * 16 + l15) * LDT + half * 8];
            a[i].h[0] = *(const v8bf*)p;
            a[i].h[1] = *(const v8bf*)(p + 16);
        }
        // B fragments: lane<16 holds K[0..7],K[16..23]; lane>=16 K[8..15],K[24..31] (col = l15)
        FragU bfr[2];
        #pragma unroll
        for (int j = 0; j < 2; ++j) {
            const bf16* p = &Blds[cur][(wx * 32 + j * 16 + l15) * LDT + half * 8];
            bfr[j].h[0] = *(const v8bf*)p;
            bfr[j].h[1] = *(const v8bf*)(p + 16);
        }
        #pragma unroll
        for (int i = 0; i < 4; ++i)
            #pragma unroll
            for (int j = 0; j < 2; ++j)
                acc[i][j] = __builtin_amdgcn_wmma_f32_16x16x32_bf16(
                    false, a[i].v, false, bfr[j].v, (short)0, acc[i][j], false, false);

#if USE_ASYNC_LDS
        if (more) __builtin_amdgcn_s_wait_asynccnt(0);
#else
        if (more) {
            *(v8bf*)&Alds[nxt][r0 * LDT + q * 8] = ra0;
            *(v8bf*)&Alds[nxt][r1 * LDT + q * 8] = ra1;
            *(v8bf*)&Blds[nxt][r0 * LDT + q * 8] = rb0;
            *(v8bf*)&Blds[nxt][r1 * LDT + q * 8] = rb1;
        }
#endif
        __syncthreads();
    }

    // epilogue: acc[i][j][p] -> row = m0+wy*64+i*16+p+8*half, col = n0+wx*32+j*16+l15
    #pragma unroll
    for (int i = 0; i < 4; ++i) {
        #pragma unroll
        for (int j = 0; j < 2; ++j) {
            const int col = n0 + wx * 32 + j * 16 + l15;
            const float bv = bias[col];
            #pragma unroll
            for (int p = 0; p < 8; ++p) {
                const int row = m0 + wy * 64 + i * 16 + p + 8 * half;
                const size_t idx = (size_t)row * Hd + col;
                float v = acc[i][j][p] + bv;
                if (MODE == 0) {
                    hbuf[idx] = v;
                    comb_bf[(size_t)row * (2 * Hd) + Hd + col] = f2bf(v);
                } else if (MODE == 1) {
                    zbuf[idx] = 1.0f / (1.0f + __expf(-v));
                } else if (MODE == 2) {
                    float r = 1.0f / (1.0f + __expf(-v));
                    rh_bf[idx] = f2bf(r * hbuf[idx]);
                } else {
                    float ht = tanhf(v);
                    float zv = zbuf[idx];
                    float hn = (1.0f - zv) * hbuf[idx] + zv * ht;
                    hbuf[idx] = hn;
                    comb_bf[(size_t)row * (2 * Hd) + Hd + col] = f2bf(hn);
                }
            }
        }
    }
}

// ---------------- per-step decode: logits, gumbel-ST token, prev, outputs ----
__global__ __launch_bounds__(256)
void decode_step(const float* __restrict__ h, const float* __restrict__ W_out,
                 const float* __restrict__ b_out, const float* __restrict__ u_noise,
                 const float* __restrict__ W_tok, const float* __restrict__ b_tok,
                 bf16* __restrict__ comb_bf, float* __restrict__ rc,
                 float* __restrict__ nt, float* __restrict__ out_msg,
                 float* __restrict__ out_logits, float* __restrict__ out_nt, int t) {
    const int lane = threadIdx.x & 31;
    const int wave = threadIdx.x >> 5;
    const int b = blockIdx.x * 8 + wave;       // one wave per batch row
    const float* hrow = h + (size_t)b * Hd;

    float acc[Vn] = {};
    for (int c = lane; c < Hd; c += 32) {
        float hv = hrow[c];
        #pragma unroll
        for (int v = 0; v < Vn; ++v) acc[v] += hv * W_out[c * Vn + v];
    }
    #pragma unroll
    for (int v = 0; v < Vn; ++v)
        #pragma unroll
        for (int off = 16; off > 0; off >>= 1)
            acc[v] += __shfl_xor(acc[v], off, 32);
    // all lanes now hold full sums
    float logits[Vn], yl[Vn];
    float mx = -3.0e38f;
    #pragma unroll
    for (int v = 0; v < Vn; ++v) {
        logits[v] = acc[v] + b_out[v];
        float u = u_noise[(size_t)t * Bn * Vn + (size_t)b * Vn + v];
        float g = -__logf(-__logf(u + 1e-10f) + 1e-10f);
        yl[v] = logits[v] + g;                 // TAU == 1
        mx = fmaxf(mx, yl[v]);
    }
    float e[Vn], s = 0.0f;
    #pragma unroll
    for (int v = 0; v < Vn; ++v) { e[v] = __expf(yl[v] - mx); s += e[v]; }
    float inv = 1.0f / s;
    int amax = 0; float best = yl[0];
    #pragma unroll
    for (int v = 1; v < Vn; ++v) if (yl[v] > best) { best = yl[v]; amax = v; }

    float rc_old = rc[b];
    float token[Vn];
    #pragma unroll
    for (int v = 0; v < Vn; ++v) {
        float ys = e[v] * inv;
        token[v] = ((v == amax ? 1.0f : 0.0f) - ys) + ys;   // straight-through
    }
    if (lane < Vn) {
        out_msg[(size_t)b * (Tn * Vn) + t * Vn + lane] = token[lane] * rc_old;
        out_logits[(size_t)t * Bn * Vn + (size_t)b * Vn + lane] = logits[lane];
    }
    if (lane == 0) {
        float ntn = nt[b] + rc_old;
        nt[b] = ntn;
        rc[b] = rc_old * (1.0f - token[Vn - 1]);            // STOP = last symbol
        if (t == Tn - 1) out_nt[b] = ntn;
    }
    // prev = token @ W_tok + b_tok  -> comb_bf[:, 0:H] (bf16)
    bf16* crow = comb_bf + (size_t)b * (2 * Hd);
    for (int c = lane; c < Hd; c += 32) {
        float p = b_tok[c];
        #pragma unroll
        for (int v = 0; v < Vn; ++v) p += token[v] * W_tok[(size_t)v * Hd + c];
        crow[c] = f2bf(p);
    }
}

// ---------------- host launcher ----------------
extern "C" void kernel_launch(void* const* d_in, const int* in_sizes, int n_in,
                              void* d_out, int out_size, void* d_ws, size_t ws_size,
                              hipStream_t stream) {
    const float* x      = (const float*)d_in[0];
    const float* u_nse  = (const float*)d_in[1];
    const float* W_enc  = (const float*)d_in[2];
    const float* b_enc  = (const float*)d_in[3];
    const float* start  = (const float*)d_in[4];
    const float* W_tok  = (const float*)d_in[5];
    const float* b_tok  = (const float*)d_in[6];
    const float* Wz     = (const float*)d_in[7];
    const float* bz     = (const float*)d_in[8];
    const float* Wr     = (const float*)d_in[9];
    const float* br     = (const float*)d_in[10];
    const float* Wh     = (const float*)d_in[11];
    const float* bh     = (const float*)d_in[12];
    const float* W_out  = (const float*)d_in[13];
    const float* b_out  = (const float*)d_in[14];

    float* out_msg    = (float*)d_out;
    float* out_logits = out_msg + (size_t)Bn * Tn * Vn;
    float* out_nt     = out_logits + (size_t)Tn * Bn * Vn;

    // workspace carve-up
    char* p = (char*)d_ws;
    auto take = [&](size_t bytes) { char* q = p; p += (bytes + 255) & ~size_t(255); return q; };
    bf16*  x_bf   = (bf16*) take((size_t)Bn * DIN * 2);
    bf16*  WencT  = (bf16*) take((size_t)Hd * Kd * 2);
    bf16*  WzT    = (bf16*) take((size_t)Hd * Kd * 2);
    bf16*  WrT    = (bf16*) take((size_t)Hd * Kd * 2);
    bf16*  WhT    = (bf16*) take((size_t)Hd * Kd * 2);
    float* hbuf   = (float*)take((size_t)Bn * Hd * 4);
    float* zbuf   = (float*)take((size_t)Bn * Hd * 4);
    bf16*  comb   = (bf16*) take((size_t)Bn * 2 * Hd * 2);
    bf16*  rh     = (bf16*) take((size_t)Bn * Hd * 2);
    float* rc     = (float*)take((size_t)Bn * 4);
    float* nt     = (float*)take((size_t)Bn * 4);

    // one-time conversions (per call; deterministic)
    k_cvt_bf16<<<(Bn * DIN) / 256, 256, 0, stream>>>(x, x_bf, Bn * DIN);
    dim3 tb(32, 8), tg(Kd / 32, Hd / 32);
    k_transpose_bf16<<<tg, tb, 0, stream>>>(W_enc, WencT, Kd, Hd);
    k_transpose_bf16<<<tg, tb, 0, stream>>>(Wz,    WzT,   Kd, Hd);
    k_transpose_bf16<<<tg, tb, 0, stream>>>(Wr,    WrT,   Kd, Hd);
    k_transpose_bf16<<<tg, tb, 0, stream>>>(Wh,    WhT,   Kd, Hd);
    k_init<<<(Bn * Hd) / 256, 256, 0, stream>>>(start, comb, rc, nt);

    dim3 gg(Bn / 128, Hd / 128);   // 64 x 8 workgroups
    // encoder: h0 = x @ W_enc + b_enc
    gemm_bf16<0><<<gg, 256, 0, stream>>>(x_bf, DIN, x_bf, DIN, Kd, WencT, Kd,
                                         b_enc, hbuf, zbuf, comb, rh);
    for (int t = 0; t < Tn; ++t) {
        gemm_bf16<1><<<gg, 256, 0, stream>>>(comb, 2 * Hd, comb, 2 * Hd, Kd, WzT, Kd,
                                             bz, hbuf, zbuf, comb, rh);
        gemm_bf16<2><<<gg, 256, 0, stream>>>(comb, 2 * Hd, comb, 2 * Hd, Kd, WrT, Kd,
                                             br, hbuf, zbuf, comb, rh);
        gemm_bf16<3><<<gg, 256, 0, stream>>>(comb, 2 * Hd, rh, Hd, Hd, WhT, Kd,
                                             bh, hbuf, zbuf, comb, rh);
        decode_step<<<Bn / 8, 256, 0, stream>>>(hbuf, W_out, b_out, u_nse, W_tok,
                                                b_tok, comb, rc, nt, out_msg,
                                                out_logits, out_nt, t);
    }
}
